// LatentODEDecoder_32650341384389
// MI455X (gfx1250) — matrix operand order repaired and checked by reading, not compiled, not run on hardware
//
#include <hip/hip_runtime.h>

typedef __attribute__((ext_vector_type(16))) _Float16 v16h;
typedef __attribute__((ext_vector_type(8)))  _Float16 v8h;
typedef __attribute__((ext_vector_type(8)))  float    v8f;

#define DIM   256
#define TT    64
#define NSUB  2          // RK4 substeps per eval interval (h = 1/128)

// Build a 16-half WMMA fragment from two 16-byte LDS runs.
__device__ __forceinline__ v16h ld2_frag(const _Float16* p0, const _Float16* p1) {
  v8h lo = *(const v8h*)p0;
  v8h hi = *(const v8h*)p1;
  return __builtin_shufflevector(lo, hi, 0,1,2,3,4,5,6,7,8,9,10,11,12,13,14,15);
}

__global__ void __launch_bounds__(256, 1)
latent_ode_kernel(const float* __restrict__ h0,
                  const float* __restrict__ t_eval,
                  const float* __restrict__ W1,
                  const float* __restrict__ b1,
                  const float* __restrict__ W2,
                  const float* __restrict__ b2,
                  float* __restrict__ out)
{
  extern __shared__ _Float16 smem[];
  _Float16* sW1 = smem;              // [256][256] f16, row-major == B-fragment layout
  _Float16* sW2 = smem + 65536;      // [256][256]
  _Float16* sIn = smem + 131072;     // [32][256]  vf stage input (A source)
  _Float16* sY  = smem + 139264;     // [32][256]  hidden activation (A source)

  const int tid = threadIdx.x;

  // Cooperative f32 -> f16 weight load into LDS (one-time).
  for (int i = tid; i < DIM * DIM; i += 256) {
    sW1[i] = (_Float16)W1[i];
    sW2[i] = (_Float16)W2[i];
  }
  __syncthreads();

  const int lane = tid & 31;
  const int wave = tid >> 5;
  const int lr   = lane & 15;        // column-within-tile / A row
  const int hi   = lane >> 4;        // lane half
  const int slab = wave >> 2;        // which 16-row slab (0..1)
  const int jn0  = (wave & 3) * 4;   // first of this wave's 4 N-tiles
  const int R0   = blockIdx.x * 32;  // workgroup's batch-row base

  // Bias values for this wave's columns (same for all 8 C-rows).
  float bv1[4], bv2[4];
#pragma unroll
  for (int j = 0; j < 4; ++j) {
    int col = (jn0 + j) * 16 + lr;
    bv1[j] = b1[col];
    bv2[j] = b2[col];
  }

  // State in f32 registers, C-matrix layout: x[j][r] = state[row r+8*hi][col 16*(jn0+j)+lr]
  v8f x[4];
#pragma unroll
  for (int j = 0; j < 4; ++j) {
    int col = (jn0 + j) * 16 + lr;
#pragma unroll
    for (int r = 0; r < 8; ++r) {
      int grow = R0 + slab * 16 + r + 8 * hi;
      x[j][r] = h0[(size_t)grow * DIM + col];
    }
  }

  const int aRow = slab * 16 + lr;       // A-fragment row in sIn/sY
  const int aK0  = hi ? 8 : 0;           // A: lanes>=16 hold K+8 / K+24
  const int bK0  = hi ? 8 : 0;           // B: lanes>=16 hold K+8 / K+24

  v8f k[4], acc[4];
  float tprev = 0.0f;

  for (int ti = 0; ti < TT; ++ti) {
    float tcur  = t_eval[ti];
    float hstep = (tcur - tprev) / (float)NSUB;
    tprev = tcur;

    for (int sub = 0; sub < NSUB; ++sub) {
#pragma unroll 1
      for (int stage = 0; stage < 4; ++stage) {
        float a = (stage == 3) ? hstep : 0.5f * hstep;

        __syncthreads();   // protect sIn/sY against WAR from previous stage
        // ---- scatter stage input xt = x + a*k (f32 regs -> f16 A-layout in LDS)
#pragma unroll
        for (int j = 0; j < 4; ++j) {
          int col = (jn0 + j) * 16 + lr;
#pragma unroll
          for (int r = 0; r < 8; ++r) {
            float v = (stage == 0) ? x[j][r] : x[j][r] + a * k[j][r];
            sIn[(slab * 16 + r + 8 * hi) * DIM + col] = (_Float16)v;
          }
        }
        __syncthreads();

        // ---- GEMM1: hidden = relu(xt @ W1^T + b1) -> sY (f16)
        v16h afr[8];
#pragma unroll
        for (int kc = 0; kc < 8; ++kc) {
          const _Float16* p = &sIn[aRow * DIM + kc * 32 + aK0];
          afr[kc] = ld2_frag(p, p + 16);
        }
#pragma unroll
        for (int j = 0; j < 4; ++j) {
          int n = (jn0 + j) * 16 + lr;
          v8f c;
#pragma unroll
          for (int r = 0; r < 8; ++r) c[r] = bv1[j];
#pragma unroll
          for (int kc = 0; kc < 8; ++kc) {
            const _Float16* wp = &sW1[n * DIM + kc * 32 + bK0];
            v16h bfr = ld2_frag(wp, wp + 16);
            c = __builtin_amdgcn_wmma_f32_16x16x32_f16(
                    false, afr[kc], false, bfr, (short)0, c, false, false);
          }
          int col = (jn0 + j) * 16 + lr;
#pragma unroll
          for (int r = 0; r < 8; ++r) {
            float v = c[r] > 0.0f ? c[r] : 0.0f;   // ReLU
            sY[(slab * 16 + r + 8 * hi) * DIM + col] = (_Float16)v;
          }
        }
        __syncthreads();

        // ---- GEMM2: k = hidden @ W2^T + b2 (stays in f32 registers)
#pragma unroll
        for (int kc = 0; kc < 8; ++kc) {
          const _Float16* p = &sY[aRow * DIM + kc * 32 + aK0];
          afr[kc] = ld2_frag(p, p + 16);
        }
#pragma unroll
        for (int j = 0; j < 4; ++j) {
          int n = (jn0 + j) * 16 + lr;
          v8f c;
#pragma unroll
          for (int r = 0; r < 8; ++r) c[r] = bv2[j];
#pragma unroll
          for (int kc = 0; kc < 8; ++kc) {
            const _Float16* wp = &sW2[n * DIM + kc * 32 + bK0];
            v16h bfr = ld2_frag(wp, wp + 16);
            c = __builtin_amdgcn_wmma_f32_16x16x32_f16(
                    false, afr[kc], false, bfr, (short)0, c, false, false);
          }
          k[j] = c;
        }

        // ---- RK4 accumulate: acc = k1 + 2k2 + 2k3 + k4
        float w = (stage == 1 || stage == 2) ? 2.0f : 1.0f;
#pragma unroll
        for (int j = 0; j < 4; ++j)
#pragma unroll
          for (int r = 0; r < 8; ++r)
            acc[j][r] = (stage == 0) ? k[j][r] : acc[j][r] + w * k[j][r];
      }
      // x += h/6 * acc
      float s = hstep * (1.0f / 6.0f);
#pragma unroll
      for (int j = 0; j < 4; ++j)
#pragma unroll
        for (int r = 0; r < 8; ++r)
          x[j][r] += s * acc[j][r];
    }

    // ---- emit snapshot at t_eval[ti]: out[b][ti][dim]
#pragma unroll
    for (int j = 0; j < 4; ++j) {
      int col = (jn0 + j) * 16 + lr;
#pragma unroll
      for (int r = 0; r < 8; ++r) {
        int grow = R0 + slab * 16 + r + 8 * hi;
        out[((size_t)grow * TT + ti) * DIM + col] = x[j][r];
      }
    }
  }
}

extern "C" void kernel_launch(void* const* d_in, const int* in_sizes, int n_in,
                              void* d_out, int out_size, void* d_ws, size_t ws_size,
                              hipStream_t stream) {
  (void)in_sizes; (void)n_in; (void)d_ws; (void)ws_size; (void)out_size;
  const float* h  = (const float*)d_in[0];
  const float* te = (const float*)d_in[1];
  const float* W1 = (const float*)d_in[2];
  const float* b1 = (const float*)d_in[3];
  const float* W2 = (const float*)d_in[4];
  const float* b2 = (const float*)d_in[5];
  float* out = (float*)d_out;

  const size_t shmem = (size_t)(65536 + 65536 + 8192 + 8192) * sizeof(_Float16); // 288 KB
  (void)hipFuncSetAttribute((const void*)latent_ode_kernel,
                            hipFuncAttributeMaxDynamicSharedMemorySize, (int)shmem);

  dim3 grid(4096 / 32);   // 128 workgroups, 32 batch rows each
  dim3 block(256);        // 8 waves: 2 row-slabs x 4 N-quarters
  latent_ode_kernel<<<grid, block, shmem, stream>>>(h, te, W1, b1, W2, b2, out);
}
